// Curv_Net_35029753266853
// MI455X (gfx1250) — matrix operand order repaired and checked by reading, not compiled, not run on hardware
//
#include <hip/hip_runtime.h>

// ---------------------------------------------------------------------------
// Curv-Net forward, fused for MI455X (gfx1250, wave32, WMMA).
//
// Heavy path: Y = sigmoid(X @ (W*mask)^T + b), 3 big + 2 small layers as bf16
// WMMA (v_wmma_f32_16x16x32_bf16), ~277 GFLOP -> compute-bound at bf16 rate.
// W tiles double-buffered via GLOBAL_LOAD_ASYNC_TO_LDS_B128 (ASYNCcnt), so
// tile i+1's DMA overlaps tile i's WMMAs; X tiles converted f32->bf16 in
// VGPRs while staging. One-hot selector matmuls -> argmax-index + gather.
// ---------------------------------------------------------------------------

typedef __attribute__((ext_vector_type(16))) __bf16 v16bf;
typedef __attribute__((ext_vector_type(8)))  float  v8f;
typedef __attribute__((ext_vector_type(4)))  float  v4f;
typedef __attribute__((ext_vector_type(4)))  int    v4i;

#define BDIM 2048
#define IN_  4096
#define ED_  8192
#define PW_  2048
#define OUT_ 256
#define CL_  16
#define NK_  32
#define CAT_ 400   // OUT + CL + 4*NK

#define TM 64
#define TN 128
#define TK 32

__device__ __forceinline__ unsigned short f2bf(float f) {
    unsigned int u = __float_as_uint(f);
    unsigned int r = (u + 0x7FFFu + ((u >> 16) & 1u)) >> 16;   // RNE
    return (unsigned short)r;
}

__device__ __forceinline__ float sigmoidf(float x) {
    return 1.0f / (1.0f + __expf(-x));
}

// -------- elementwise: out_bf16 = a * m (m==nullptr -> plain convert) -------
__global__ __launch_bounds__(256)
void mask_cvt_bf16(const float* __restrict__ a, const float* __restrict__ m,
                   unsigned short* __restrict__ out, size_t n) {
    size_t i = ((size_t)blockIdx.x * 256 + threadIdx.x) * 4;
    if (i >= n) return;
    #pragma unroll
    for (int j = 0; j < 4; ++j) {
        float v = a[i + j];
        if (m) v *= m[i + j];
        out[i + j] = f2bf(v);
    }
}

// -------- one-hot selector -> column indices --------------------------------
__global__ void onehot_idx(const float* __restrict__ mask, int* __restrict__ idx,
                           int nrows) {
    size_t t = (size_t)blockIdx.x * blockDim.x + threadIdx.x;
    size_t total = (size_t)nrows * NK_;
    if (t >= total) return;
    if (mask[t] > 0.5f) idx[t % NK_] = (int)(t / NK_);   // unique writer per col
}

// -------- kept[b, base+k] = src[b, idx[k]] ----------------------------------
__global__ void gather_cols(const float* __restrict__ src, const int* __restrict__ idx,
                            float* __restrict__ kept, int rowlen, int base) {
    int t = blockIdx.x * blockDim.x + threadIdx.x;
    if (t >= BDIM * NK_) return;
    int b = t >> 5, k = t & 31;
    kept[(size_t)b * (4 * NK_) + base + k] = src[(size_t)b * rowlen + idx[k]];
}

// -------- act = other*(c1*c3) + act*(c2*c3), coeffs per column --------------
__global__ __launch_bounds__(256)
void mix_kernel(float* __restrict__ act, const float* __restrict__ other,
                const float* __restrict__ c1, const float* __restrict__ c2,
                const float* __restrict__ c3, int rowlen) {
    int col = blockIdx.x * 256 + threadIdx.x;
    int row = blockIdx.y;
    size_t i = (size_t)row * rowlen + col;
    float ca = c1[col] * c3[col], cb = c2[col] * c3[col];
    act[i] = other[i] * ca + act[i] * cb;
}

// ---------------------------------------------------------------------------
// Main WMMA GEMM: Y[Bdim,N] = post( sigmoid( X[Bdim,K] @ W[N,K]^T + bias ) )
// Double-buffered: W tile via async global->LDS DMA (2 b128/lane, ASYNCcnt),
// X tile f32->bf16 in VGPRs. Pipeline depth 2; s_wait_asynccnt 0x2 retires
// the older tile while the newer streams in, overlapping DMA with WMMA.
// Block = 256 thr (8 waves), tile 64x128, K-step 32, wave tile 32x32 (2x2).
// ---------------------------------------------------------------------------
template<bool HAS_PS>
__global__ __launch_bounds__(256)
void gemm_bf16_wmma(const float* __restrict__ X, const unsigned short* __restrict__ W,
                    const float* __restrict__ bias, const float* __restrict__ postscale,
                    float* __restrict__ Y, int N, int K) {
    __shared__ __align__(16) unsigned short Xs[2][TM * TK];   // 2 x 4 KB
    __shared__ __align__(16) unsigned short Ws[2][TN * TK];   // 2 x 8 KB

    const int tid  = threadIdx.x;
    const int lane = tid & 31;
    const int wave = tid >> 5;
    const int wm   = wave >> 2;      // 0..1  (M)
    const int wn   = wave & 3;       // 0..3  (N)
    const int blockM = blockIdx.y * TM;
    const int blockN = blockIdx.x * TN;

    v8f acc[2][2];
    #pragma unroll
    for (int i = 0; i < 2; ++i)
        #pragma unroll
        for (int j = 0; j < 2; ++j)
            #pragma unroll
            for (int e = 0; e < 8; ++e) acc[i][j][e] = 0.0f;

    // staging maps: X tile 64x32 f32 -> 8 floats/thread; W tile 128x32 bf16 -> 16/thread
    const int xr = tid >> 2, xc = (tid & 3) * 8;
    const int wr = tid >> 1, wc = (tid & 1) * 16;
    const float*          Xp = X + (size_t)(blockM + xr) * K + xc;
    const unsigned short* Wp = W + (size_t)(blockN + wr) * K + wc;
    // LDS byte addresses of this thread's W staging slots (flat low 32 == LDS offset)
    const unsigned wlds0 = (unsigned)(size_t)&Ws[0][wr * TK + wc];
    const unsigned wlds1 = (unsigned)(size_t)&Ws[1][wr * TK + wc];

    // LDS fragment addresses (CDNA5 16-bit A 16x32 / B 32x16 lane layouts)
    const int arow = wm * 32 + (lane & 15);
    const int akb  = (lane >> 4) * 8;          // A: K 0-7 / 8-15, +16 for hi half
    const int bcol = wn * 32 + (lane & 15);
    const int bkb  = (lane >> 4) * 8;          // B: K 0-7 / 8-15, +16 for hi half

    union Frag { v16bf v; v4i q[2]; };

    auto stage = [&](int buf) {
        // async DMA: W tile slice, 32 B/lane; offset applies to LDS & global addr
        unsigned wl = buf ? wlds1 : wlds0;
        asm volatile("global_load_async_to_lds_b128 %0, %1, off"
                     :: "v"(wl), "v"(Wp) : "memory");
        asm volatile("global_load_async_to_lds_b128 %0, %1, off offset:16"
                     :: "v"(wl), "v"(Wp) : "memory");
        // X slice: f32 -> bf16 through VGPRs
        v4f x0 = *(const v4f*)(Xp + 0);
        v4f x1 = *(const v4f*)(Xp + 4);
        union { unsigned short h[8]; v4i q; } cv;
        #pragma unroll
        for (int j = 0; j < 4; ++j) { cv.h[j] = f2bf(x0[j]); cv.h[4 + j] = f2bf(x1[j]); }
        *(v4i*)&Xs[buf][xr * TK + xc] = cv.q;
        Xp += TK; Wp += TK;
    };

    auto compute = [&](int buf) {
        Frag a0, a1, b0, b1;
        const unsigned short* xb = Xs[buf];
        const unsigned short* wb = Ws[buf];
        a0.q[0] = *(const v4i*)&xb[(arow +  0) * TK + akb];
        a0.q[1] = *(const v4i*)&xb[(arow +  0) * TK + akb + 16];
        a1.q[0] = *(const v4i*)&xb[(arow + 16) * TK + akb];
        a1.q[1] = *(const v4i*)&xb[(arow + 16) * TK + akb + 16];
        b0.q[0] = *(const v4i*)&wb[(bcol +  0) * TK + bkb];
        b0.q[1] = *(const v4i*)&wb[(bcol +  0) * TK + bkb + 16];
        b1.q[0] = *(const v4i*)&wb[(bcol + 16) * TK + bkb];
        b1.q[1] = *(const v4i*)&wb[(bcol + 16) * TK + bkb + 16];
        acc[0][0] = __builtin_amdgcn_wmma_f32_16x16x32_bf16(false, a0.v, false, b0.v,
                                                            (short)0, acc[0][0], false, false);
        acc[0][1] = __builtin_amdgcn_wmma_f32_16x16x32_bf16(false, a0.v, false, b1.v,
                                                            (short)0, acc[0][1], false, false);
        acc[1][0] = __builtin_amdgcn_wmma_f32_16x16x32_bf16(false, a1.v, false, b0.v,
                                                            (short)0, acc[1][0], false, false);
        acc[1][1] = __builtin_amdgcn_wmma_f32_16x16x32_bf16(false, a1.v, false, b1.v,
                                                            (short)0, acc[1][1], false, false);
    };

    const int niter = K / TK;        // even, >= 8 for all layers
    stage(0);                        // tile 0 -> buf 0
    stage(1);                        // tile 1 -> buf 1

    int i = 0;
    for (; i + 2 < niter; i += 2) {
        // older tile's 2 DMAs retired; newer tile's 2 remain in flight
        asm volatile("s_wait_asynccnt 0x2" ::: "memory");
        __syncthreads();             // cross-wave visibility (dscnt wait + barrier)
        compute(0);
        __syncthreads();             // all waves done reading buf0
        stage(i + 2);                // tile i+2 -> buf0, overlaps next compute

        asm volatile("s_wait_asynccnt 0x2" ::: "memory");
        __syncthreads();
        compute(1);
        __syncthreads();
        stage(i + 3);                // tile i+3 -> buf1
    }
    // tail: tiles niter-2 (buf0) and niter-1 (buf1), nothing left to stage
    asm volatile("s_wait_asynccnt 0x2" ::: "memory");
    __syncthreads();
    compute(0);
    __syncthreads();
    asm volatile("s_wait_asynccnt 0x0" ::: "memory");
    __syncthreads();
    compute(1);

    // epilogue: C/D layout (VGPR j -> M = j + 8*(lane>=16), N = lane&15)
    #pragma unroll
    for (int mi = 0; mi < 2; ++mi) {
        #pragma unroll
        for (int ni = 0; ni < 2; ++ni) {
            int col = blockN + wn * 32 + ni * 16 + (lane & 15);
            float bb  = bias[col];
            float psc = HAS_PS ? postscale[col] : 1.0f;
            #pragma unroll
            for (int j = 0; j < 8; ++j) {
                int row = blockM + wm * 32 + mi * 16 + (lane >> 4) * 8 + j;
                Y[(size_t)row * N + col] = sigmoidf(acc[mi][ni][j] + bb) * psc;
            }
        }
    }
}

// -------- concat [act5 | kept(128) | clinn] -> xcat [B,400] -----------------
__global__ void concat_kernel(const float* __restrict__ act5, const float* __restrict__ kept,
                              const float* __restrict__ clinn, float* __restrict__ xcat) {
    int t = blockIdx.x * blockDim.x + threadIdx.x;
    if (t >= BDIM * CAT_) return;
    int b = t / CAT_, c = t % CAT_;
    float v;
    if (c < OUT_)            v = act5[(size_t)b * OUT_ + c];
    else if (c < OUT_ + 128) v = kept[(size_t)b * 128 + (c - OUT_)];
    else                     v = clinn[(size_t)b * CL_ + (c - OUT_ - 128)];
    xcat[t] = v;
}

// -------- head: lp = sig(xcat@W6^T); lp -= mean(lp); out = lp@W7^T ----------
__global__ __launch_bounds__(256)
void head_kernel(const float* __restrict__ xcat, const float* __restrict__ W6,
                 const float* __restrict__ W7, float* __restrict__ out) {
    __shared__ float xrow[CAT_];
    __shared__ float red[256];
    int b = blockIdx.x, t = threadIdx.x;
    for (int i = t; i < CAT_; i += 256) xrow[i] = xcat[(size_t)b * CAT_ + i];
    __syncthreads();
    const float* w = W6 + (size_t)t * CAT_;
    float s = 0.0f;
    for (int i = 0; i < CAT_; ++i) s += xrow[i] * w[i];
    float v = sigmoidf(s);
    red[t] = v; __syncthreads();
    for (int off = 128; off > 0; off >>= 1) { if (t < off) red[t] += red[t + off]; __syncthreads(); }
    float mean = red[0] * (1.0f / 256.0f);
    __syncthreads();
    red[t] = (v - mean) * W7[t]; __syncthreads();
    for (int off = 128; off > 0; off >>= 1) { if (t < off) red[t] += red[t + off]; __syncthreads(); }
    if (t == 0) out[b] = red[0];
}

// ---------------------------------------------------------------------------
extern "C" void kernel_launch(void* const* d_in, const int* in_sizes, int n_in,
                              void* d_out, int out_size, void* d_ws, size_t ws_size,
                              hipStream_t stream) {
    const float* x_gene   = (const float*)d_in[0];
    const float* x_invmea = (const float*)d_in[1];
    const float* x_curv   = (const float*)d_in[2];
    const float* clinn    = (const float*)d_in[3];
    const float* Adj      = (const float*)d_in[4];
    const float* edge_m   = (const float*)d_in[5];
    const float* path_m   = (const float*)d_in[6];
    const float* tg_mask  = (const float*)d_in[7];
    const float* ti_mask  = (const float*)d_in[8];
    const float* tc_mask  = (const float*)d_in[9];
    const float* tp_mask  = (const float*)d_in[10];
    const float* W1 = (const float*)d_in[11]; const float* b1 = (const float*)d_in[12];
    const float* W2 = (const float*)d_in[13]; const float* b2 = (const float*)d_in[14];
    const float* W3 = (const float*)d_in[15]; const float* b3 = (const float*)d_in[16];
    const float* W4 = (const float*)d_in[17]; const float* b4 = (const float*)d_in[18];
    const float* W5 = (const float*)d_in[19]; const float* b5 = (const float*)d_in[20];
    const float* W6 = (const float*)d_in[21]; const float* W7 = (const float*)d_in[22];
    const float* mp11 = (const float*)d_in[23]; const float* mp12 = (const float*)d_in[24];
    const float* mp1  = (const float*)d_in[25];
    const float* mp21 = (const float*)d_in[26]; const float* mp22 = (const float*)d_in[27];
    const float* mp2  = (const float*)d_in[28]; const float* mp3  = (const float*)d_in[29];
    float* out = (float*)d_out;

    // workspace layout (256B aligned)
    char* base = (char*)d_ws;
    size_t ofs = 0;
    auto alloc = [&](size_t bytes) { char* p = base + ofs; ofs = (ofs + bytes + 255) & ~(size_t)255; return p; };
    unsigned short* W1m = (unsigned short*)alloc((size_t)IN_ * IN_ * 2);
    unsigned short* W2m = (unsigned short*)alloc((size_t)ED_ * IN_ * 2);
    unsigned short* W3m = (unsigned short*)alloc((size_t)PW_ * ED_ * 2);
    unsigned short* W4m = (unsigned short*)alloc((size_t)OUT_ * PW_ * 2);
    unsigned short* W5m = (unsigned short*)alloc((size_t)OUT_ * OUT_ * 2);
    float* act1 = (float*)alloc((size_t)BDIM * IN_ * 4);
    float* act2 = (float*)alloc((size_t)BDIM * ED_ * 4);
    float* act3 = (float*)alloc((size_t)BDIM * PW_ * 4);
    float* act4 = (float*)alloc((size_t)BDIM * OUT_ * 4);
    float* act5 = (float*)alloc((size_t)BDIM * OUT_ * 4);
    float* kept = (float*)alloc((size_t)BDIM * 4 * NK_ * 4);
    float* xcat = (float*)alloc((size_t)BDIM * CAT_ * 4);
    int*   idxg = (int*)alloc(NK_ * 4);
    int*   idxi = (int*)alloc(NK_ * 4);
    int*   idxc = (int*)alloc(NK_ * 4);
    int*   idxp = (int*)alloc(NK_ * 4);

    auto cvtN = [](size_t n) { return dim3((unsigned)((n / 4 + 255) / 256)); };

    // 1) one-hot selectors -> indices
    onehot_idx<<<dim3((IN_ * NK_ + 255) / 256), 256, 0, stream>>>(tg_mask, idxg, IN_);
    onehot_idx<<<dim3((IN_ * NK_ + 255) / 256), 256, 0, stream>>>(ti_mask, idxi, IN_);
    onehot_idx<<<dim3((ED_ * NK_ + 255) / 256), 256, 0, stream>>>(tc_mask, idxc, ED_);
    onehot_idx<<<dim3((PW_ * NK_ + 255) / 256), 256, 0, stream>>>(tp_mask, idxp, PW_);

    // 2) fold masks, convert weights to bf16
    mask_cvt_bf16<<<cvtN((size_t)IN_ * IN_), 256, 0, stream>>>(W1, Adj,    W1m, (size_t)IN_ * IN_);
    mask_cvt_bf16<<<cvtN((size_t)ED_ * IN_), 256, 0, stream>>>(W2, edge_m, W2m, (size_t)ED_ * IN_);
    mask_cvt_bf16<<<cvtN((size_t)PW_ * ED_), 256, 0, stream>>>(W3, path_m, W3m, (size_t)PW_ * ED_);
    mask_cvt_bf16<<<cvtN((size_t)OUT_ * PW_), 256, 0, stream>>>(W4, nullptr, W4m, (size_t)OUT_ * PW_);
    mask_cvt_bf16<<<cvtN((size_t)OUT_ * OUT_), 256, 0, stream>>>(W5, nullptr, W5m, (size_t)OUT_ * OUT_);

    // 3) kept_gene from raw x_gene
    gather_cols<<<dim3(BDIM * NK_ / 256), 256, 0, stream>>>(x_gene, idxg, kept, IN_, 0);

    dim3 blk(256);
    // 4) layer1: act1 = sig(x_gene @ (W1*Adj)^T + b1)
    gemm_bf16_wmma<false><<<dim3(IN_ / TN, BDIM / TM), blk, 0, stream>>>(x_gene, W1m, b1, nullptr, act1, IN_, IN_);
    // 5) mix with x_invmea
    mix_kernel<<<dim3(IN_ / 256, BDIM), blk, 0, stream>>>(act1, x_invmea, mp11, mp12, mp1, IN_);
    // 6) kept_invmea
    gather_cols<<<dim3(BDIM * NK_ / 256), 256, 0, stream>>>(act1, idxi, kept, IN_, NK_);
    // 7) layer2: act2 = sig(act1 @ (W2*edge)^T + b2)
    gemm_bf16_wmma<false><<<dim3(ED_ / TN, BDIM / TM), blk, 0, stream>>>(act1, W2m, b2, nullptr, act2, ED_, IN_);
    // 8) mix with x_curv
    mix_kernel<<<dim3(ED_ / 256, BDIM), blk, 0, stream>>>(act2, x_curv, mp21, mp22, mp2, ED_);
    // 9) kept_curv
    gather_cols<<<dim3(BDIM * NK_ / 256), 256, 0, stream>>>(act2, idxc, kept, ED_, 2 * NK_);
    // 10) layer3: act3 = sig(act2 @ (W3*path)^T + b3) * mp3  (mp3 fused as postscale)
    gemm_bf16_wmma<true><<<dim3(PW_ / TN, BDIM / TM), blk, 0, stream>>>(act2, W3m, b3, mp3, act3, PW_, ED_);
    // 11) kept_path (from mp3-scaled act3, matching reference order)
    gather_cols<<<dim3(BDIM * NK_ / 256), 256, 0, stream>>>(act3, idxp, kept, PW_, 3 * NK_);
    // 12) layer4 + 13) layer5
    gemm_bf16_wmma<false><<<dim3(OUT_ / TN, BDIM / TM), blk, 0, stream>>>(act3, W4m, b4, nullptr, act4, OUT_, PW_);
    gemm_bf16_wmma<false><<<dim3(OUT_ / TN, BDIM / TM), blk, 0, stream>>>(act4, W5m, b5, nullptr, act5, OUT_, OUT_);
    // 14) concat, 15) head
    concat_kernel<<<dim3((BDIM * CAT_ + 255) / 256), 256, 0, stream>>>(act5, kept, clinn, xcat);
    head_kernel<<<dim3(BDIM), blk, 0, stream>>>(xcat, W6, W7, out);
}
